// ChannelHyperedgeInteraction_55757265437216
// MI455X (gfx1250) — matrix operand order repaired and checked by reading, not compiled, not run on hardware
//
#include <hip/hip_runtime.h>
#include <hip/hip_bf16.h>
#include <math.h>

// ---------------------------------------------------------------------------
// Types for WMMA fragments (wave32, CDNA5)
// ---------------------------------------------------------------------------
typedef __attribute__((ext_vector_type(16))) _Float16 v16h;
typedef __attribute__((ext_vector_type(8)))  _Float16 v8h;
typedef __attribute__((ext_vector_type(4)))  _Float16 v4h;
typedef __attribute__((ext_vector_type(8)))  float    v8f;

union Frag16 { v16h v; v8h h[2]; };

// Problem constants
#define BB   16
#define CC   128
#define LL   256
#define DD   1024
#define HH   4096
#define BC   (BB*CC)      // 2048

// GEMM epilogue flags
#define GF_BIAS   1
#define GF_GELU   2
#define GF_OUTF32 4
#define GF_VTRANS 8   // store f16 output transposed per 128-row batch: [b][n][m%128]

// ---------------------------------------------------------------------------
// 1) Masked mean-pool over L: E[bc,d], Eh[bc,d] (f16), total[bc]
//    Streams the 2.15 GB x tensor once with float4 loads (bandwidth stage).
// ---------------------------------------------------------------------------
__launch_bounds__(256)
__global__ void pool_kernel(const float* __restrict__ x, const float* __restrict__ mask,
                            float* __restrict__ E, _Float16* __restrict__ Eh,
                            float* __restrict__ total)
{
    const int bc = blockIdx.x;
    const int t  = threadIdx.x;
    const float* xrow = x + (size_t)bc * LL * DD + t * 4;
    const float* mrow = mask + (size_t)bc * LL;
    float ax = 0.f, ay = 0.f, az = 0.f, aw = 0.f, msum = 0.f;
#pragma unroll 4
    for (int l = 0; l < LL; ++l) {
        float mk = mrow[l];
        float4 xv = *(const float4*)(xrow + (size_t)l * DD);
        ax += mk * xv.x; ay += mk * xv.y; az += mk * xv.z; aw += mk * xv.w;
        msum += mk;
    }
    float inv = 1.f / fmaxf(msum, 1.f);
    ax *= inv; ay *= inv; az *= inv; aw *= inv;
    size_t o = (size_t)bc * DD + t * 4;
    float4 ev; ev.x = ax; ev.y = ay; ev.z = az; ev.w = aw;
    *(float4*)(E + o) = ev;
    v4h hv; hv[0] = (_Float16)ax; hv[1] = (_Float16)ay; hv[2] = (_Float16)az; hv[3] = (_Float16)aw;
    *(v4h*)(Eh + o) = hv;
    if (t == 0) total[bc] = msum;
}

// ---------------------------------------------------------------------------
// 2) Weight convert fp32 [K,N] -> f16 transposed [N,K] (LDS-tiled transpose)
// ---------------------------------------------------------------------------
__launch_bounds__(256)
__global__ void convT_kernel(const float* __restrict__ W, _Float16* __restrict__ Wt,
                             int K, int N)
{
    __shared__ float tile[32][33];
    const int kb = blockIdx.y * 32, nb = blockIdx.x * 32;
    const int tx = threadIdx.x & 31, ty = threadIdx.x >> 5;  // 32x8
#pragma unroll
    for (int i = 0; i < 32; i += 8)
        tile[ty + i][tx] = W[(size_t)(kb + ty + i) * N + nb + tx];
    __syncthreads();
#pragma unroll
    for (int i = 0; i < 32; i += 8)
        Wt[(size_t)(nb + ty + i) * K + kb + tx] = (_Float16)tile[tx][ty + i];
}

// ---------------------------------------------------------------------------
// 3) Generic WMMA GEMM: C[M,N] = A[M,K] @ Bt[N,K]^T (+bias)(+GELU)
//    256 threads = 8 waves; block tile 64(M) x 128(N); wave tile 16x64.
//    A fragment:  a[j]   = A[m][kb16+j], kb16=(lane>>4)*16 (contiguous 16)
//    B fragment:  b[j]   = Bt[n][kb16+j], kb16=(lane>>4)*16 (contiguous 16)
//    D fragment:  c[r]   = D[(lane>>4)*8 + r][lane&15]
// ---------------------------------------------------------------------------
__launch_bounds__(256)
__global__ void wmma_gemm_kernel(const _Float16* __restrict__ A,
                                 const _Float16* __restrict__ Bt,
                                 const float* __restrict__ bias,
                                 float* __restrict__ Cf, _Float16* __restrict__ Ch,
                                 int M, int N, int K,
                                 size_t strideA, size_t strideB, size_t strideC,
                                 int flags)
{
    const int lane  = threadIdx.x & 31;
    const int wave  = threadIdx.x >> 5;      // 0..7
    const int waveM = wave & 3;              // 4 waves along M
    const int waveN = wave >> 2;             // 2 waves along N
    const int z     = blockIdx.z;

    const int mBase  = blockIdx.y * 64 + waveM * 16;
    const int nBase0 = blockIdx.x * 128 + waveN * 64;
    if (mBase >= M || nBase0 >= N) return;   // wave-uniform, EXEC stays full

    const _Float16* Ab = A  + (size_t)z * strideA;
    const _Float16* Bb = Bt + (size_t)z * strideB;

    const int lidx  = lane & 15;
    const int half  = lane >> 4;
    const int koffA = half * 16;
    const int koffB = half * 16;

    const _Float16* aRow = Ab + (size_t)(mBase + lidx) * K + koffA;
    const _Float16* bRow0 = Bb + (size_t)(nBase0 + 0 * 16 + lidx) * K + koffB;
    const _Float16* bRow1 = Bb + (size_t)(nBase0 + 1 * 16 + lidx) * K + koffB;
    const _Float16* bRow2 = Bb + (size_t)(nBase0 + 2 * 16 + lidx) * K + koffB;
    const _Float16* bRow3 = Bb + (size_t)(nBase0 + 3 * 16 + lidx) * K + koffB;

    v8f acc0 = {}, acc1 = {}, acc2 = {}, acc3 = {};

    for (int k0 = 0; k0 < K; k0 += 32) {
        Frag16 a, b0, b1, b2, b3;
        a.h[0]  = *(const v8h*)(aRow + k0);
        a.h[1]  = *(const v8h*)(aRow + k0 + 8);
        b0.h[0] = *(const v8h*)(bRow0 + k0);
        b0.h[1] = *(const v8h*)(bRow0 + k0 + 8);
        b1.h[0] = *(const v8h*)(bRow1 + k0);
        b1.h[1] = *(const v8h*)(bRow1 + k0 + 8);
        b2.h[0] = *(const v8h*)(bRow2 + k0);
        b2.h[1] = *(const v8h*)(bRow2 + k0 + 8);
        b3.h[0] = *(const v8h*)(bRow3 + k0);
        b3.h[1] = *(const v8h*)(bRow3 + k0 + 8);
        acc0 = __builtin_amdgcn_wmma_f32_16x16x32_f16(false, a.v, false, b0.v, (short)0, acc0, false, false);
        acc1 = __builtin_amdgcn_wmma_f32_16x16x32_f16(false, a.v, false, b1.v, (short)0, acc1, false, false);
        acc2 = __builtin_amdgcn_wmma_f32_16x16x32_f16(false, a.v, false, b2.v, (short)0, acc2, false, false);
        acc3 = __builtin_amdgcn_wmma_f32_16x16x32_f16(false, a.v, false, b3.v, (short)0, acc3, false, false);
    }

    const int mRow0 = mBase + half * 8;
    v8f accs[4] = { acc0, acc1, acc2, acc3 };
#pragma unroll
    for (int t = 0; t < 4; ++t) {
        const int n = nBase0 + t * 16 + lidx;
        const float bval = (flags & GF_BIAS) ? bias[n] : 0.f;
        if (flags & GF_VTRANS) {
            // f16 transposed store: out[(m/128)][n][m%128], contiguous over r
            v8h pack;
#pragma unroll
            for (int r = 0; r < 8; ++r)
                pack[r] = (_Float16)(accs[t][r] + bval);
            const int mG = mRow0;
            size_t addr = (size_t)(mG / 128) * ((size_t)N * 128) + (size_t)n * 128 + (mG % 128);
            *(v8h*)(Ch + addr) = pack;
        } else {
#pragma unroll
            for (int r = 0; r < 8; ++r) {
                float xv = accs[t][r] + bval;
                if (flags & GF_GELU)
                    xv = 0.5f * xv * (1.f + erff(xv * 0.70710678118654752f));
                size_t addr = (size_t)z * strideC + (size_t)(mRow0 + r) * N + n;
                if (flags & GF_OUTF32) Cf[addr] = xv;
                else                   Ch[addr] = (_Float16)xv;
            }
        }
    }
}

// ---------------------------------------------------------------------------
// 4) Fused mask-overlap modulation + softmax over k (128), attn -> f16
// ---------------------------------------------------------------------------
__launch_bounds__(128)
__global__ void softmax_kernel(const float* __restrict__ S, const float* __restrict__ mask,
                               const float* __restrict__ total, _Float16* __restrict__ attn)
{
    __shared__ float mrow[LL];
    __shared__ float red[128];
    const int bc = blockIdx.x;
    const int b  = bc >> 7;
    const int kc = threadIdx.x;
    mrow[kc]       = mask[(size_t)bc * LL + kc];
    mrow[kc + 128] = mask[(size_t)bc * LL + 128 + kc];
    __syncthreads();
    const float* mk = mask + (size_t)(b * CC + kc) * LL;
    float joint = 0.f;
#pragma unroll 4
    for (int l = 0; l < LL; ++l) joint += mrow[l] * mk[l];
    float tp = fmaxf(total[bc] + total[b * CC + kc], 1.f);
    float factor = 0.5f + joint / tp;                 // 0.5 + 0.5 * (2*joint/tp)
    float s = S[(size_t)bc * CC + kc] * 0.03125f * factor;  // 1/sqrt(1024)
    // max reduce
    red[kc] = s; __syncthreads();
    for (int off = 64; off > 0; off >>= 1) {
        if (kc < off) red[kc] = fmaxf(red[kc], red[kc + off]);
        __syncthreads();
    }
    float mx = red[0]; __syncthreads();
    float e = expf(s - mx);
    red[kc] = e; __syncthreads();
    for (int off = 64; off > 0; off >>= 1) {
        if (kc < off) red[kc] += red[kc + off];
        __syncthreads();
    }
    float inv = 1.f / red[0];
    attn[(size_t)bc * CC + kc] = (_Float16)(e * inv);
}

// ---------------------------------------------------------------------------
// 5) Residual + LayerNorm: out = LN(X+Y)*g + be ; optional f16 copy
// ---------------------------------------------------------------------------
__launch_bounds__(256)
__global__ void ln_kernel(const float* __restrict__ X, const float* __restrict__ Y,
                          const float* __restrict__ g, const float* __restrict__ be,
                          float* __restrict__ out, _Float16* __restrict__ outh)
{
    __shared__ float s1[256], s2[256];
    const int bc = blockIdx.x, t = threadIdx.x;
    const size_t base = (size_t)bc * DD + t * 4;
    float4 xv = *(const float4*)(X + base);
    float4 yv = *(const float4*)(Y + base);
    float v0 = xv.x + yv.x, v1 = xv.y + yv.y, v2 = xv.z + yv.z, v3 = xv.w + yv.w;
    s1[t] = v0 + v1 + v2 + v3;
    s2[t] = v0 * v0 + v1 * v1 + v2 * v2 + v3 * v3;
    __syncthreads();
    for (int off = 128; off > 0; off >>= 1) {
        if (t < off) { s1[t] += s1[t + off]; s2[t] += s2[t + off]; }
        __syncthreads();
    }
    const float mean = s1[0] * (1.f / DD);
    const float var  = s2[0] * (1.f / DD) - mean * mean;
    const float r    = rsqrtf(var + 1e-5f);
    float4 gv  = *(const float4*)(g  + t * 4);
    float4 bev = *(const float4*)(be + t * 4);
    float4 ov;
    ov.x = (v0 - mean) * r * gv.x + bev.x;
    ov.y = (v1 - mean) * r * gv.y + bev.y;
    ov.z = (v2 - mean) * r * gv.z + bev.z;
    ov.w = (v3 - mean) * r * gv.w + bev.w;
    *(float4*)(out + base) = ov;
    if (outh) {
        v4h hv; hv[0] = (_Float16)ov.x; hv[1] = (_Float16)ov.y;
                hv[2] = (_Float16)ov.z; hv[3] = (_Float16)ov.w;
        *(v4h*)(outh + base) = hv;
    }
}

// ---------------------------------------------------------------------------
// Launcher
// ---------------------------------------------------------------------------
extern "C" void kernel_launch(void* const* d_in, const int* in_sizes, int n_in,
                              void* d_out, int out_size, void* d_ws, size_t ws_size,
                              hipStream_t stream)
{
    (void)in_sizes; (void)n_in; (void)out_size; (void)ws_size;
    const float* x    = (const float*)d_in[0];
    const float* mask = (const float*)d_in[1];
    const float* Wq = (const float*)d_in[2];  const float* bq = (const float*)d_in[3];
    const float* Wk = (const float*)d_in[4];  const float* bk = (const float*)d_in[5];
    const float* Wv = (const float*)d_in[6];  const float* bv = (const float*)d_in[7];
    const float* Wo = (const float*)d_in[8];  const float* bo = (const float*)d_in[9];
    const float* W1 = (const float*)d_in[10]; const float* b1 = (const float*)d_in[11];
    const float* W2 = (const float*)d_in[12]; const float* b2 = (const float*)d_in[13];
    const float* g1 = (const float*)d_in[14]; const float* be1 = (const float*)d_in[15];
    const float* g2 = (const float*)d_in[16]; const float* be2 = (const float*)d_in[17];
    float* out = (float*)d_out;

    char* ws = (char*)d_ws;
    size_t off = 0;
    auto alloc = [&](size_t bytes) -> char* {
        char* p = ws + off;
        off += (bytes + 255) & ~(size_t)255;
        return p;
    };
    float*    E     = (float*)   alloc((size_t)BC * DD * 4);
    _Float16* Eh    = (_Float16*)alloc((size_t)BC * DD * 2);
    float*    total = (float*)   alloc((size_t)BC * 4);
    _Float16* Wqt   = (_Float16*)alloc((size_t)DD * DD * 2);
    _Float16* Wkt   = (_Float16*)alloc((size_t)DD * DD * 2);
    _Float16* Wvt   = (_Float16*)alloc((size_t)DD * DD * 2);
    _Float16* Wot   = (_Float16*)alloc((size_t)DD * DD * 2);
    _Float16* W1t   = (_Float16*)alloc((size_t)DD * HH * 2);
    _Float16* W2t   = (_Float16*)alloc((size_t)HH * DD * 2);
    _Float16* qh    = (_Float16*)alloc((size_t)BC * DD * 2);
    _Float16* kh    = (_Float16*)alloc((size_t)BC * DD * 2);
    _Float16* vT    = (_Float16*)alloc((size_t)BC * DD * 2);  // [b][d][c]
    float*    Sbuf  = (float*)   alloc((size_t)BB * CC * CC * 4);
    _Float16* attn  = (_Float16*)alloc((size_t)BB * CC * CC * 2);
    _Float16* ctx   = (_Float16*)alloc((size_t)BC * DD * 2);
    float*    tmp32 = (float*)   alloc((size_t)BC * DD * 4);  // attn-proj out, then FFN out
    float*    E2    = (float*)   alloc((size_t)BC * DD * 4);
    _Float16* E2h   = (_Float16*)alloc((size_t)BC * DD * 2);
    _Float16* hbuf  = (_Float16*)alloc((size_t)BC * HH * 2);

    // 1) masked mean-pool (bandwidth-dominant: streams 2.15 GB of x)
    pool_kernel<<<BC, 256, 0, stream>>>(x, mask, E, Eh, total);

    // 2) weight casts + transpose to [N,K] f16
    convT_kernel<<<dim3(DD/32, DD/32), 256, 0, stream>>>(Wq, Wqt, DD, DD);
    convT_kernel<<<dim3(DD/32, DD/32), 256, 0, stream>>>(Wk, Wkt, DD, DD);
    convT_kernel<<<dim3(DD/32, DD/32), 256, 0, stream>>>(Wv, Wvt, DD, DD);
    convT_kernel<<<dim3(DD/32, DD/32), 256, 0, stream>>>(Wo, Wot, DD, DD);
    convT_kernel<<<dim3(HH/32, DD/32), 256, 0, stream>>>(W1, W1t, DD, HH);
    convT_kernel<<<dim3(DD/32, HH/32), 256, 0, stream>>>(W2, W2t, HH, DD);

    // 3) q,k,v = E @ W + b  (M=2048,N=1024,K=1024); v stored transposed per batch
    dim3 gQKV(DD/128, BC/64, 1);
    wmma_gemm_kernel<<<gQKV, 256, 0, stream>>>(Eh, Wqt, bq, nullptr, qh,
        BC, DD, DD, 0, 0, 0, GF_BIAS);
    wmma_gemm_kernel<<<gQKV, 256, 0, stream>>>(Eh, Wkt, bk, nullptr, kh,
        BC, DD, DD, 0, 0, 0, GF_BIAS);
    wmma_gemm_kernel<<<gQKV, 256, 0, stream>>>(Eh, Wvt, bv, nullptr, vT,
        BC, DD, DD, 0, 0, 0, GF_BIAS | GF_VTRANS);

    // 4) S[b] = q[b] @ k[b]^T  (batched: M=128,N=128,K=1024)
    dim3 gS(CC/128, CC/64, BB);
    wmma_gemm_kernel<<<gS, 256, 0, stream>>>(qh, kh, nullptr, Sbuf, nullptr,
        CC, CC, DD, (size_t)CC*DD, (size_t)CC*DD, (size_t)CC*CC, GF_OUTF32);

    // 5) overlap modulation + softmax
    softmax_kernel<<<BC, 128, 0, stream>>>(Sbuf, mask, total, attn);

    // 6) ctx[b] = attn[b] @ v[b]  (batched: M=128,N=1024,K=128; B = vT rows)
    dim3 gCtx(DD/128, CC/64, BB);
    wmma_gemm_kernel<<<gCtx, 256, 0, stream>>>(attn, vT, nullptr, nullptr, ctx,
        CC, DD, CC, (size_t)CC*CC, (size_t)DD*CC, (size_t)CC*DD, 0);

    // 7) attn output projection: tmp32 = ctx @ Wo + bo (fp32 out)
    wmma_gemm_kernel<<<gQKV, 256, 0, stream>>>(ctx, Wot, bo, tmp32, nullptr,
        BC, DD, DD, 0, 0, 0, GF_BIAS | GF_OUTF32);

    // 8) E2 = LN(E + attn_out)
    ln_kernel<<<BC, 256, 0, stream>>>(E, tmp32, g1, be1, E2, E2h);

    // 9) h = gelu(E2 @ W1 + b1)  (M=2048,N=4096,K=1024), f16 out
    dim3 gF1(HH/128, BC/64, 1);
    wmma_gemm_kernel<<<gF1, 256, 0, stream>>>(E2h, W1t, b1, nullptr, hbuf,
        BC, HH, DD, 0, 0, 0, GF_BIAS | GF_GELU);

    // 10) tmp32 = h @ W2 + b2  (M=2048,N=1024,K=4096), fp32 out
    dim3 gF2(DD/128, BC/64, 1);
    wmma_gemm_kernel<<<gF2, 256, 0, stream>>>(hbuf, W2t, b2, tmp32, nullptr,
        BC, DD, HH, 0, 0, 0, GF_BIAS | GF_OUTF32);

    // 11) out = LN(E2 + ffn_out)
    ln_kernel<<<BC, 256, 0, stream>>>(E2, tmp32, g2, be2, out, nullptr);
}
